// GraphSCModule_10943576670842
// MI455X (gfx1250) — matrix operand order repaired and checked by reading, not compiled, not run on hardware
//
#include <hip/hip_runtime.h>
#include <math.h>

typedef __attribute__((ext_vector_type(16))) _Float16 v16h;
typedef __attribute__((ext_vector_type(8)))  _Float16 v8h;
typedef __attribute__((ext_vector_type(8)))  float    v8f;

union V16U { v16h v; v8h h[2]; };
union PK2 { _Float16 h[2]; unsigned int u; };

#define DEV __device__ __forceinline__

DEV v8f wmma32(v16h a, v16h b, v8f c) {
  // v_wmma_f32_16x16x32_f16: D = A(16x32) * B(32x16) + C
  return __builtin_amdgcn_wmma_f32_16x16x32_f16(false, a, false, b, (short)0, c, false, false);
}

// A fragment (16x32 f16) from row-major [ld] matrix, top-left (row0,col0).
// Lane l: m = l&15; halves[0..7] = cols col0+8*(l>>4)..+7 ; halves[8..15] = +16.
DEV v16h load_A(const _Float16* base, int ld, int row0, int col0, int lane) {
  int ln = lane & 15, lg = lane >> 4;
  const _Float16* p = base + (size_t)(row0 + ln) * ld + col0 + 8 * lg;
  V16U u; u.h[0] = *(const v8h*)p; u.h[1] = *(const v8h*)(p + 16);
  return u.v;
}
// B fragment (32x16 f16) where B[k,n] = Wrow[n0+n, k0+k] (i.e. B = Wrow^T tile).
// Lane l: n = l&15; halves[0..7] = Wrow[n0+n, k0 + 8*(l>>4) + j]; halves[8..15] = +16.
DEV v16h load_BT(const _Float16* base, int ld, int n0, int k0, int lane) {
  int ln = lane & 15, lg = lane >> 4;
  const _Float16* p = base + (size_t)(n0 + ln) * ld + k0 + 8 * lg;
  V16U u; u.h[0] = *(const v8h*)p; u.h[1] = *(const v8h*)(p + 16);
  return u.v;
}

// ---------------------------------------------------------------- utilities
__global__ void k_zero(float* p, long long n) {
  long long i = (long long)blockIdx.x * blockDim.x + threadIdx.x;
  if (i < n) p[i] = 0.f;
}

__global__ void k_f32_to_f16(const float* __restrict__ s, _Float16* __restrict__ d, int n) {
  int i = blockIdx.x * blockDim.x + threadIdx.x;
  if (i < n) d[i] = (_Float16)s[i];
}

__global__ __launch_bounds__(256) void k_means(const float* __restrict__ s,
                                               const float* __restrict__ t,
                                               float* __restrict__ means, int C) {
  __shared__ float red[256];
  int tid = threadIdx.x;
  float acc[6] = {0.f, 0.f, 0.f, 0.f, 0.f, 0.f};
  for (int c = tid; c < C; c += 256) {
    acc[0] += s[3 * c + 0]; acc[1] += s[3 * c + 1]; acc[2] += s[3 * c + 2];
    acc[3] += t[3 * c + 0]; acc[4] += t[3 * c + 1]; acc[5] += t[3 * c + 2];
  }
  for (int k = 0; k < 6; k++) {
    red[tid] = acc[k]; __syncthreads();
    for (int off = 128; off > 0; off >>= 1) {
      if (tid < off) red[tid] += red[tid + off];
      __syncthreads();
    }
    if (tid == 0) means[k] = red[0] / (float)C;
    __syncthreads();
  }
}

// ------------------------------------------------------------------- stem
__global__ __launch_bounds__(128) void k_conv1(
    const float* __restrict__ src, const float* __restrict__ tgt,
    const float* __restrict__ means, const float* __restrict__ W,
    const float* __restrict__ B, float* __restrict__ xout, float* __restrict__ gstat) {
  int c = blockIdx.x, t = threadIdx.x;
  __shared__ float emb[128];
  __shared__ float red[128], red2[128];
  if (t < 126) {
    int half = t / 63, rem = t - half * 63, coord = rem / 21, j = rem - coord * 21;
    const float* P = half ? tgt : src;
    float x = P[c * 3 + coord] - means[half * 3 + coord];
    float v;
    if (j == 0) v = x;
    else if (j <= 10) v = sinf(x * (float)(1 << (j - 1)));
    else v = cosf(x * (float)(1 << (j - 11)));
    emb[t] = v;
  }
  __syncthreads();
  float y = B[t];
  const float* wr = W + t * 126;
#pragma unroll 6
  for (int i = 0; i < 126; i++) y += wr[i] * emb[i];
  xout[(size_t)c * 128 + t] = y;
  red[t] = y; red2[t] = y * y;
  __syncthreads();
  if (t < 8) {
    float sum = 0.f, ss = 0.f;
#pragma unroll
    for (int i = 0; i < 16; i++) { sum += red[t * 16 + i]; ss += red2[t * 16 + i]; }
    atomicAdd(&gstat[t], sum);
    atomicAdd(&gstat[8 + t], ss);
  }
}

__global__ void k_gnfin(const float* __restrict__ gstat, float* __restrict__ gfin, float inv_n) {
  int t = threadIdx.x;
  if (t < 8) {
    float mu = gstat[t] * inv_n;
    float var = gstat[8 + t] * inv_n - mu * mu;
    gfin[t] = mu;
    gfin[8 + t] = rsqrtf(var + 1e-5f);
  }
}

// GN apply (+leaky) then 128x128 conv; optional GN partials for the next GN.
__global__ __launch_bounds__(128) void k_convmid(
    const float* __restrict__ xin, const float* __restrict__ gfin,
    const float* __restrict__ gg, const float* __restrict__ gb,
    const float* __restrict__ W, const float* __restrict__ B,
    float* __restrict__ xout, float* gstat) {
  int c = blockIdx.x, t = threadIdx.x;
  __shared__ float xr[128];
  __shared__ float red[128], red2[128];
  int g = t >> 4;
  float v = xin[(size_t)c * 128 + t];
  v = (v - gfin[g]) * gfin[8 + g] * gg[t] + gb[t];
  v = v > 0.f ? v : 0.01f * v;
  xr[t] = v;
  __syncthreads();
  float y = B[t];
  const float* wr = W + t * 128;
#pragma unroll 8
  for (int i = 0; i < 128; i++) y += wr[i] * xr[i];
  xout[(size_t)c * 128 + t] = y;
  if (gstat) {
    red[t] = y; red2[t] = y * y;
    __syncthreads();
    if (t < 8) {
      float sum = 0.f, ss = 0.f;
#pragma unroll
      for (int i = 0; i < 16; i++) { sum += red[t * 16 + i]; ss += red2[t * 16 + i]; }
      atomicAdd(&gstat[t], sum);
      atomicAdd(&gstat[8 + t], ss);
    }
  }
}

// -------------------------------------------------- spatial consistency (f16)
__global__ __launch_bounds__(128) void k_sc(
    const float* __restrict__ src, const float* __restrict__ tgt,
    const float* __restrict__ means, const int* __restrict__ idx,
    _Float16* __restrict__ sc) {
  int m = blockIdx.x, t = threadIdx.x;
  __shared__ float ps[128][3], pt[128][3];
  int ci = idx[m * 128 + t];
#pragma unroll
  for (int d = 0; d < 3; d++) {
    ps[t][d] = src[ci * 3 + d] - means[d];
    pt[t][d] = tgt[ci * 3 + d] - means[3 + d];
  }
  __syncthreads();
  float a0 = ps[t][0], a1 = ps[t][1], a2 = ps[t][2];
  float b0 = pt[t][0], b1 = pt[t][1], b2 = pt[t][2];
  _Float16* out = sc + ((size_t)m * 128 + t) * 128;
  for (int j = 0; j < 128; j++) {
    float dx = a0 - ps[j][0], dy = a1 - ps[j][1], dz = a2 - ps[j][2];
    float ds = sqrtf(fmaxf(dx * dx + dy * dy + dz * dz, 0.f));
    dx = b0 - pt[j][0]; dy = b1 - pt[j][1]; dz = b2 - pt[j][2];
    float dt = sqrtf(fmaxf(dx * dx + dy * dy + dz * dz, 0.f));
    float del = fabsf(ds - dt);
    float v = 1.f - del * del * 100.f;  // 1/sigma_d^2 = 100
    out[j] = (_Float16)fmaxf(v, 0.f);
  }
}

// ------------------------------------------------------ weights / seg-sum
__global__ void k_sumw(const int* __restrict__ idx, const float* __restrict__ w,
                       float* __restrict__ sumw, int n) {
  int t = blockIdx.x * blockDim.x + threadIdx.x;
  if (t < n) atomicAdd(&sumw[idx[t]], w[t]);
}
__global__ void k_normw(const int* __restrict__ idx, const float* __restrict__ w,
                        const float* __restrict__ sumw, float* __restrict__ nw, int n) {
  int t = blockIdx.x * blockDim.x + threadIdx.x;
  if (t < n) nw[t] = w[t] / (sumw[idx[t]] + 1e-8f);
}

__global__ void k_gather(const float* __restrict__ cf, const int* __restrict__ idx,
                         _Float16* __restrict__ lf) {
  int t = blockIdx.x * blockDim.x + threadIdx.x;  // one (m,k) row per thread
  int c = idx[t];
  const float4* s = (const float4*)(cf + (size_t)c * 128);
  v8h* d = (v8h*)(lf + (size_t)t * 128);
#pragma unroll
  for (int i = 0; i < 16; i++) {
    float4 a = s[2 * i], b = s[2 * i + 1];
    v8h h = {(_Float16)a.x, (_Float16)a.y, (_Float16)a.z, (_Float16)a.w,
             (_Float16)b.x, (_Float16)b.y, (_Float16)b.z, (_Float16)b.w};
    d[i] = h;
  }
}

__global__ void k_scatter(const _Float16* __restrict__ lf, const int* __restrict__ idx,
                          const float* __restrict__ nw, float* __restrict__ cf) {
  int t = blockIdx.x * blockDim.x + threadIdx.x;
  int rc = t >> 5;       // (m*K+k)
  int q4 = t & 31;       // 4-channel chunk
  int c = idx[rc];
  float w = nw[rc];
  const _Float16* s = lf + (size_t)rc * 128 + q4 * 4;
  float* d = cf + (size_t)c * 128 + q4 * 4;
#pragma unroll
  for (int i = 0; i < 4; i++) atomicAdd(d + i, (float)s[i] * w);
}

// --------------------------------------------------- attention (per m,head)
__global__ __launch_bounds__(128) void k_attn(
    const _Float16* __restrict__ lf, const _Float16* __restrict__ sc,
    const int* __restrict__ mask, _Float16* __restrict__ attn,
    const _Float16* __restrict__ Wq, const float* __restrict__ Bq,
    const _Float16* __restrict__ Wk, const float* __restrict__ Bk,
    const _Float16* __restrict__ Wv, const float* __restrict__ Bvb) {
  const int m = blockIdx.x, h = blockIdx.y;
  const int tid = threadIdx.x, wave = tid >> 5, lane = tid & 31;
  const int ln = lane & 15, lg = lane >> 4;
  __shared__ _Float16 sX[128 * 128];   // x, later reused for attention probs a
  __shared__ _Float16 sQ[128 * 32];
  __shared__ _Float16 sK[128 * 32];
  __shared__ _Float16 sVT[32 * 128];   // v transposed (d-major)

  __builtin_prefetch(Wk, 0, 3);
  __builtin_prefetch(Wv, 0, 3);

  {  // stage x tile: thread tid copies row tid (256B)
    const v8h* s = (const v8h*)(lf + (size_t)m * 128 * 128 + (size_t)tid * 128);
    v8h* d = (v8h*)(sX + tid * 128);
#pragma unroll
    for (int i = 0; i < 16; i++) d[i] = s[i];
  }
  __syncthreads();

  // q,k,v projections: each wave does 4 output tiles; A loaded once, three
  // independent WMMA chains (q,k,v) share it and pipeline the matrix unit.
  const int hb = h * 32;
  for (int tt = wave; tt < 16; tt += 4) {
    int tm = tt >> 1, tn = tt & 1;
    v16h a[4];
#pragma unroll
    for (int kk = 0; kk < 4; kk++) a[kk] = load_A(sX, 128, 16 * tm, 32 * kk, lane);
    v8f aq = {}, ak = {}, av = {};
#pragma unroll
    for (int kk = 0; kk < 4; kk++) {
      aq = wmma32(a[kk], load_BT(Wq, 128, hb + 16 * tn, 32 * kk, lane), aq);
      ak = wmma32(a[kk], load_BT(Wk, 128, hb + 16 * tn, 32 * kk, lane), ak);
      av = wmma32(a[kk], load_BT(Wv, 128, hb + 16 * tn, 32 * kk, lane), av);
    }
    float bq = Bq[hb + 16 * tn + ln];
    float bk = Bk[hb + 16 * tn + ln];
    float bv = Bvb[hb + 16 * tn + ln];
    int col = 16 * tn + ln;
#pragma unroll
    for (int r = 0; r < 8; r++) {
      int row = 16 * tm + r + 8 * lg;
      sQ[row * 32 + col] = (_Float16)(aq[r] + bq);
      sK[row * 32 + col] = (_Float16)(ak[r] + bk);
    }
    int vrow0 = 16 * tm + 8 * lg;  // consecutive r -> consecutive LDS halves
#pragma unroll
    for (int r = 0; r < 8; r += 2) {
      PK2 pk;
      pk.h[0] = (_Float16)(av[r] + bv);
      pk.h[1] = (_Float16)(av[r + 1] + bv);
      *(unsigned int*)(sVT + col * 128 + vrow0 + r) = pk.u;
    }
  }
  __syncthreads();

  // scores: wave owns rows [32w,32w+32) — 2x8 tiles, dh=32 -> one WMMA per tile
  v8f sacc[2][8];
  const _Float16* scp = sc + (size_t)m * 128 * 128;
  const int* mp = mask + m * 128;
  bool okv[8];
#pragma unroll
  for (int tc = 0; tc < 8; tc++) okv[tc] = mp[16 * tc + ln] != 0;
  const float scale = 0.17677669529663687f;  // 1/sqrt(32)
#pragma unroll
  for (int tr = 0; tr < 2; tr++) {
    int row0 = 32 * wave + 16 * tr;
    v16h aq = load_A(sQ, 32, row0, 0, lane);
#pragma unroll
    for (int tc = 0; tc < 8; tc++) {
      v16h bk = load_BT(sK, 32, 16 * tc, 0, lane);
      v8f acc = {};
      acc = wmma32(aq, bk, acc);
      int j = 16 * tc + ln;
#pragma unroll
      for (int r = 0; r < 8; r++) {
        int i = row0 + r + 8 * lg;
        float s = acc[r] * scale * (float)scp[(size_t)i * 128 + j];
        acc[r] = okv[tc] ? s : -__builtin_inff();
      }
      sacc[tr][tc] = acc;
    }
  }
  // softmax per row: half-wave shuffles (rows live in 16-lane halves)
#pragma unroll
  for (int tr = 0; tr < 2; tr++) {
#pragma unroll
    for (int r = 0; r < 8; r++) {
      float mx = -__builtin_inff();
#pragma unroll
      for (int tc = 0; tc < 8; tc++) mx = fmaxf(mx, sacc[tr][tc][r]);
#pragma unroll
      for (int off = 1; off < 16; off <<= 1) mx = fmaxf(mx, __shfl_xor(mx, off, 32));
      float sum = 0.f;
#pragma unroll
      for (int tc = 0; tc < 8; tc++) {
        float e = __expf(sacc[tr][tc][r] - mx);
        sacc[tr][tc][r] = e; sum += e;
      }
#pragma unroll
      for (int off = 1; off < 16; off <<= 1) sum += __shfl_xor(sum, off, 32);
      float inv = 1.f / sum;
#pragma unroll
      for (int tc = 0; tc < 8; tc++) sacc[tr][tc][r] *= inv;
    }
  }
  // write probs into sX (own rows only)
#pragma unroll
  for (int tr = 0; tr < 2; tr++) {
    int row0 = 32 * wave + 16 * tr;
#pragma unroll
    for (int tc = 0; tc < 8; tc++) {
#pragma unroll
      for (int r = 0; r < 8; r++)
        sX[(row0 + r + 8 * lg) * 128 + 16 * tc + ln] = (_Float16)sacc[tr][tc][r];
    }
  }
  __syncthreads();

  // o = a @ v : output 128x32 for this head; A loaded once per row-tile,
  // two parallel accumulator chains for the two d-column tiles.
  _Float16* ao = attn + (size_t)m * 128 * 128;
#pragma unroll
  for (int tr = 0; tr < 2; tr++) {
    int row0 = 32 * wave + 16 * tr;
    v16h a[4];
#pragma unroll
    for (int kk = 0; kk < 4; kk++) a[kk] = load_A(sX, 128, row0, 32 * kk, lane);
    v8f o0 = {}, o1 = {};
#pragma unroll
    for (int kk = 0; kk < 4; kk++) {
      o0 = wmma32(a[kk], load_BT(sVT, 128, 0, 32 * kk, lane), o0);
      o1 = wmma32(a[kk], load_BT(sVT, 128, 16, 32 * kk, lane), o1);
    }
#pragma unroll
    for (int r = 0; r < 8; r++) {
      size_t rowoff = (size_t)(row0 + r + 8 * lg) * 128 + hb + ln;
      ao[rowoff] = (_Float16)o0[r];
      ao[rowoff + 16] = (_Float16)o1[r];
    }
  }
}

// -------------------- O-proj + LN1 + FFN + LN2 fused (per patch, 8 waves) ---
__global__ __launch_bounds__(256) void k_fuse(
    _Float16* __restrict__ lf, const _Float16* __restrict__ attn,
    const _Float16* __restrict__ Wo, const float* __restrict__ Bo,
    const _Float16* __restrict__ W1, const float* __restrict__ B1,
    const _Float16* __restrict__ W2, const float* __restrict__ B2,
    const float* __restrict__ G1, const float* __restrict__ Be1,
    const float* __restrict__ G2, const float* __restrict__ Be2) {
  const int m = blockIdx.x, tid = threadIdx.x, wave = tid >> 5, lane = tid & 31;
  const int ln = lane & 15, lg = lane >> 4;
  const int row0 = 16 * wave;  // each wave owns a disjoint 16-row band
  __shared__ _Float16 sX1[128 * 128];
  __shared__ _Float16 sH[128 * 128];
  const _Float16* ag = attn + (size_t)m * 128 * 128;
  _Float16* xg = lf + (size_t)m * 128 * 128;

  __builtin_prefetch(W1, 0, 3);
  __builtin_prefetch(W2, 0, 3);

  // O projection + residual (A fragments loaded once, reused for all 8 tiles)
  v8f y[8];
  {
    v16h a4[4];
#pragma unroll
    for (int kk = 0; kk < 4; kk++) a4[kk] = load_A(ag, 128, row0, 32 * kk, lane);
#pragma unroll
    for (int tc = 0; tc < 8; tc++) {
      v8f acc = {};
#pragma unroll
      for (int kk = 0; kk < 4; kk++)
        acc = wmma32(a4[kk], load_BT(Wo, 128, 16 * tc, 32 * kk, lane), acc);
      float bias = Bo[16 * tc + ln];
#pragma unroll
      for (int r = 0; r < 8; r++) {
        int i = row0 + r + 8 * lg;
        acc[r] += bias + (float)xg[(size_t)i * 128 + 16 * tc + ln];
      }
      y[tc] = acc;
    }
  }
  // LN1 (row-local in half-wave) -> y = x1, also staged as f16 in sX1
  float g1v[8], b1v[8];
#pragma unroll
  for (int tc = 0; tc < 8; tc++) { g1v[tc] = G1[16 * tc + ln]; b1v[tc] = Be1[16 * tc + ln]; }
#pragma unroll
  for (int r = 0; r < 8; r++) {
    float s = 0.f, ss = 0.f;
#pragma unroll
    for (int tc = 0; tc < 8; tc++) { float v = y[tc][r]; s += v; ss += v * v; }
#pragma unroll
    for (int off = 1; off < 16; off <<= 1) { s += __shfl_xor(s, off, 32); ss += __shfl_xor(ss, off, 32); }
    float mu = s * (1.f / 128.f);
    float rstd = rsqrtf(ss * (1.f / 128.f) - mu * mu + 1e-5f);
#pragma unroll
    for (int tc = 0; tc < 8; tc++) {
      float v = (y[tc][r] - mu) * rstd * g1v[tc] + b1v[tc];
      y[tc][r] = v;
      sX1[(row0 + r + 8 * lg) * 128 + 16 * tc + ln] = (_Float16)v;
    }
  }
  // FFN: hidden 256 processed as two halves of 128 to fit LDS.
  // x1 A-fragments are half-independent: load once.
  v8f f[8];
  {
    v8f z = {};
#pragma unroll
    for (int tc = 0; tc < 8; tc++) f[tc] = z;
  }
  v16h a1[4];
#pragma unroll
  for (int kk = 0; kk < 4; kk++) a1[kk] = load_A(sX1, 128, row0, 32 * kk, lane);
  for (int half = 0; half < 2; half++) {
#pragma unroll
    for (int hc = 0; hc < 8; hc++) {
      v8f acc = {};
#pragma unroll
      for (int kk = 0; kk < 4; kk++)
        acc = wmma32(a1[kk], load_BT(W1, 128, 128 * half + 16 * hc, 32 * kk, lane), acc);
      float bias = B1[128 * half + 16 * hc + ln];
#pragma unroll
      for (int r = 0; r < 8; r++) {
        float v = acc[r] + bias;
        v = v > 0.f ? v : 0.f;
        sH[(row0 + r + 8 * lg) * 128 + 16 * hc + ln] = (_Float16)v;
      }
    }
    v16h ah[4];
#pragma unroll
    for (int kk = 0; kk < 4; kk++) ah[kk] = load_A(sH, 128, row0, 32 * kk, lane);
#pragma unroll
    for (int tc = 0; tc < 8; tc++) {
      v8f acc = f[tc];
#pragma unroll
      for (int kk = 0; kk < 4; kk++)
        acc = wmma32(ah[kk], load_BT(W2, 256, 16 * tc, 128 * half + 32 * kk, lane), acc);
      f[tc] = acc;
    }
  }
  // bias + residual + LN2 + writeback (in place)
#pragma unroll
  for (int tc = 0; tc < 8; tc++) {
    float bias = B2[16 * tc + ln];
#pragma unroll
    for (int r = 0; r < 8; r++) f[tc][r] += bias + y[tc][r];
  }
  float g2v[8], b2v[8];
#pragma unroll
  for (int tc = 0; tc < 8; tc++) { g2v[tc] = G2[16 * tc + ln]; b2v[tc] = Be2[16 * tc + ln]; }
#pragma unroll
  for (int r = 0; r < 8; r++) {
    float s = 0.f, ss = 0.f;
#pragma unroll
    for (int tc = 0; tc < 8; tc++) { float v = f[tc][r]; s += v; ss += v * v; }
#pragma unroll
    for (int off = 1; off < 16; off <<= 1) { s += __shfl_xor(s, off, 32); ss += __shfl_xor(ss, off, 32); }
    float mu = s * (1.f / 128.f);
    float rstd = rsqrtf(ss * (1.f / 128.f) - mu * mu + 1e-5f);
    int i = row0 + r + 8 * lg;
#pragma unroll
    for (int tc = 0; tc < 8; tc++) {
      float v = (f[tc][r] - mu) * rstd * g2v[tc] + b2v[tc];
      xg[(size_t)i * 128 + 16 * tc + ln] = (_Float16)v;
    }
  }
}

// ------------------------------------------------------------- output head
__global__ void k_out(const float* __restrict__ cf, const float* __restrict__ ow,
                      const float* __restrict__ ob, const float* __restrict__ sumw,
                      float* __restrict__ dout, int C) {
  int c = blockIdx.x * blockDim.x + threadIdx.x;
  if (c >= C) return;
  float y = ob[0];
#pragma unroll 8
  for (int i = 0; i < 128; i++) y += cf[(size_t)c * 128 + i] * ow[i];
  dout[c] = y;
  dout[C + c] = (sumw[c] > 0.f) ? 1.f : 0.f;
}

// ---------------------------------------------------------------- launcher
extern "C" void kernel_launch(void* const* d_in, const int* in_sizes, int n_in,
                              void* d_out, int out_size, void* d_ws, size_t ws_size,
                              hipStream_t stream) {
  (void)in_sizes; (void)n_in; (void)out_size; (void)ws_size;
  constexpr int C = 30000, M = 1024, Kk = 128;

  const float* src = (const float*)d_in[0];
  const float* tgt = (const float*)d_in[1];
  const int*   idx = (const int*)d_in[2];
  const float* cw  = (const float*)d_in[3];
  const int*   msk = (const int*)d_in[4];

  // params flattened in jax pytree (sorted-key) order
  int pi = 5;
  auto F = [&](int& p) { return (const float*)d_in[p++]; };
  const float *conv1_b = F(pi), *conv1_w = F(pi);
  const float *conv2_b = F(pi), *conv2_w = F(pi);
  const float *conv3_b = F(pi), *conv3_w = F(pi);
  const float *gn1_b = F(pi), *gn1_g = F(pi), *gn2_b = F(pi), *gn2_g = F(pi);
  struct LayerP {
    const float *f1b, *f1w, *f2b, *f2w, *kb, *kw, *l1b, *l1g, *l2b, *l2g, *ob, *ow, *qb, *qw, *vb, *vw;
  } Lp[6];
  for (int l = 0; l < 6; l++) {
    Lp[l].f1b = F(pi); Lp[l].f1w = F(pi); Lp[l].f2b = F(pi); Lp[l].f2w = F(pi);
    Lp[l].kb = F(pi);  Lp[l].kw = F(pi);
    Lp[l].l1b = F(pi); Lp[l].l1g = F(pi); Lp[l].l2b = F(pi); Lp[l].l2g = F(pi);
    Lp[l].ob = F(pi);  Lp[l].ow = F(pi);  Lp[l].qb = F(pi);  Lp[l].qw = F(pi);
    Lp[l].vb = F(pi);  Lp[l].vw = F(pi);
  }
  const float *outp_b = F(pi), *outp_w = F(pi);

  // workspace carve (all 256B aligned)
  char* base = (char*)d_ws;
  size_t off = 0;
  auto carve = [&](size_t bytes) -> char* {
    char* p = base + off;
    off += bytes;
    off = (off + 255) & ~(size_t)255;
    return p;
  };
  float* means   = (float*)carve(64);
  float* gstat1  = (float*)carve(64);
  float* gfin1   = (float*)carve(64);
  float* gstat2  = (float*)carve(64);
  float* gfin2   = (float*)carve(64);
  float* xbuf    = (float*)carve((size_t)C * 128 * 4);
  float* cf      = (float*)carve((size_t)C * 128 * 4);
  float* sumw    = (float*)carve((size_t)C * 4);
  float* normw   = (float*)carve((size_t)M * Kk * 4);
  _Float16* scb  = (_Float16*)carve((size_t)M * Kk * Kk * 2);
  _Float16* lfb  = (_Float16*)carve((size_t)M * Kk * 128 * 2);
  _Float16* atb  = (_Float16*)carve((size_t)M * Kk * 128 * 2);
  _Float16* wf16 = (_Float16*)carve((size_t)6 * 131072 * 2);

  // zero accumulators (re-zeroed every call: deterministic work)
  k_zero<<<1, 64, 0, stream>>>(gstat1, 16);
  k_zero<<<1, 64, 0, stream>>>(gstat2, 16);
  k_zero<<<(C + 255) / 256, 256, 0, stream>>>(sumw, C);

  k_means<<<1, 256, 0, stream>>>(src, tgt, means, C);

  // convert per-layer weights to f16 (q,k,v,o,W1,W2)
  _Float16* wptr[6][6];
  for (int l = 0; l < 6; l++) {
    _Float16* wl = wf16 + (size_t)l * 131072;
    wptr[l][0] = wl;          wptr[l][1] = wl + 16384;  wptr[l][2] = wl + 32768;
    wptr[l][3] = wl + 49152;  wptr[l][4] = wl + 65536;  wptr[l][5] = wl + 98304;
    const float* srcs[6] = {Lp[l].qw, Lp[l].kw, Lp[l].vw, Lp[l].ow, Lp[l].f1w, Lp[l].f2w};
    const int ns[6] = {16384, 16384, 16384, 16384, 32768, 32768};
    for (int j = 0; j < 6; j++)
      k_f32_to_f16<<<(ns[j] + 255) / 256, 256, 0, stream>>>(srcs[j], wptr[l][j], ns[j]);
  }

  // stem
  k_conv1<<<C, 128, 0, stream>>>(src, tgt, means, conv1_w, conv1_b, xbuf, gstat1);
  k_gnfin<<<1, 32, 0, stream>>>(gstat1, gfin1, 1.f / (16.f * (float)C));
  k_convmid<<<C, 128, 0, stream>>>(xbuf, gfin1, gn1_g, gn1_b, conv2_w, conv2_b, xbuf, gstat2);
  k_gnfin<<<1, 32, 0, stream>>>(gstat2, gfin2, 1.f / (16.f * (float)C));
  k_convmid<<<C, 128, 0, stream>>>(xbuf, gfin2, gn2_g, gn2_b, conv3_w, conv3_b, cf, nullptr);

  // spatial-consistency weights + segment weight normalization
  k_sc<<<M, 128, 0, stream>>>(src, tgt, means, idx, scb);
  k_sumw<<<(M * Kk + 255) / 256, 256, 0, stream>>>(idx, cw, sumw, M * Kk);
  k_normw<<<(M * Kk + 255) / 256, 256, 0, stream>>>(idx, cw, sumw, normw, M * Kk);

  // transformer blocks
  for (int b = 0; b < 3; b++) {
    k_gather<<<(M * Kk) / 256, 256, 0, stream>>>(cf, idx, lfb);
    for (int l = 0; l < 2; l++) {
      int ly = b * 2 + l;
      k_attn<<<dim3(M, 4), 128, 0, stream>>>(lfb, scb, msk, atb,
          wptr[ly][0], Lp[ly].qb, wptr[ly][1], Lp[ly].kb, wptr[ly][2], Lp[ly].vb);
      k_fuse<<<M, 256, 0, stream>>>(lfb, atb,
          wptr[ly][3], Lp[ly].ob, wptr[ly][4], Lp[ly].f1b, wptr[ly][5], Lp[ly].f2b,
          Lp[ly].l1g, Lp[ly].l1b, Lp[ly].l2g, Lp[ly].l2b);
    }
    k_zero<<<(int)(((long long)C * 128 + 255) / 256), 256, 0, stream>>>(cf, (long long)C * 128);
    k_scatter<<<(M * Kk * 32) / 256, 256, 0, stream>>>(lfb, idx, normw, cf);
  }

  k_out<<<(C + 255) / 256, 256, 0, stream>>>(cf, outp_w, outp_b, sumw, (float*)d_out, C);
}
